// MultiHeadAttention_23029614641607
// MI455X (gfx1250) — matrix operand order repaired and checked by reading, not compiled, not run on hardware
//
#include <hip/hip_runtime.h>

typedef __bf16 bf16;
typedef __attribute__((ext_vector_type(16))) __bf16 v16bf;
typedef __attribute__((ext_vector_type(8)))  __bf16 v8bf;
typedef __attribute__((ext_vector_type(4)))  __bf16 v4bf;
typedef __attribute__((ext_vector_type(8)))  float  v8f;
typedef __attribute__((vector_size(16)))     int    v4i;

#define HD    64
#define NH    16
#define HDIM  1024
#define SEQ   4096
#define BSZ   2
#define MTOT  (BSZ * SEQ)   // 8192

#define LDS_AS __attribute__((address_space(3)))
#define GLB_AS __attribute__((address_space(1)))

// ---------------------------------------------------------------------------
// helpers
// ---------------------------------------------------------------------------
static __device__ __forceinline__ v16bf combine8(v8bf lo, v8bf hi) {
  v16bf r;
#pragma unroll
  for (int i = 0; i < 8; ++i) { r[i] = lo[i]; r[8 + i] = hi[i]; }
  return r;
}

static __device__ __forceinline__ v8f wmma_bf16(v16bf a, v16bf b, v8f c) {
  // D = A(16x32 bf16) * B(32x16 bf16) + C(16x16 f32)
  return __builtin_amdgcn_wmma_f32_16x16x32_bf16(
      /*neg_a=*/false, a, /*neg_b=*/false, b,
      /*c_mod=*/(short)0, c, /*reuse_a=*/false, /*reuse_b=*/false);
}

// async 16-byte copy global -> LDS (ASYNCcnt-tracked)
static __device__ __forceinline__ void async_copy_b128(const bf16* gsrc,
                                                       bf16* ldst) {
#if __has_builtin(__builtin_amdgcn_global_load_async_to_lds_b128)
  __builtin_amdgcn_global_load_async_to_lds_b128(
      (GLB_AS v4i*)gsrc, (LDS_AS v4i*)ldst, 0, 0);
#else
  unsigned loff = (unsigned)(unsigned long long)(LDS_AS bf16*)ldst;
  asm volatile("global_load_async_to_lds_b128 %0, %1, off"
               :
               : "v"(loff), "v"((unsigned long long)gsrc)
               : "memory");
#endif
}

static __device__ __forceinline__ void wait_async0() {
#if __has_builtin(__builtin_amdgcn_s_wait_asynccnt)
  __builtin_amdgcn_s_wait_asynccnt(0);
#else
  asm volatile("s_wait_asynccnt 0" ::: "memory");
#endif
}

// ---------------------------------------------------------------------------
// f32 -> bf16 conversion (vectorized x4)
// ---------------------------------------------------------------------------
__global__ __launch_bounds__(256) void cvt_f32_bf16_kernel(
    const float4* __restrict__ src, v4bf* __restrict__ dst, int n4) {
  int i = blockIdx.x * blockDim.x + threadIdx.x;
  if (i < n4) {
    float4 f = src[i];
    v4bf o;
    o[0] = (bf16)f.x; o[1] = (bf16)f.y; o[2] = (bf16)f.z; o[3] = (bf16)f.w;
    dst[i] = o;
  }
}

// ---------------------------------------------------------------------------
// QKV projection GEMM:  out = X(bf16 MxK) * W(bf16, row n holds K)^T + bias
//   vmode == 0 : store as [bh][s][d]    (Q, K)
//   vmode == 1 : store as [bh][d][s]    (V transposed, for PV B-fragments)
// grid (MTOT/64, HDIM/64), block 128 (4 waves); each wave: 16(M) x 64(N)
// ---------------------------------------------------------------------------
__global__ __launch_bounds__(128) void qkv_gemm_kernel(
    const bf16* __restrict__ X, const bf16* __restrict__ W,
    const float* __restrict__ bias, bf16* __restrict__ out, int vmode) {
  const int lane   = threadIdx.x & 31;
  const int wv     = threadIdx.x >> 5;
  const int c15    = lane & 15;
  const int half16 = (lane < 16) ? 0 : 16;
  const int base8  = (lane < 16) ? 0 : 8;
  const int m0 = (blockIdx.x * 4 + wv) * 16;
  const int n0 = blockIdx.y * 64;

  const v8f vzero = {};
  v8f acc[4];
#pragma unroll
  for (int t = 0; t < 4; ++t) acc[t] = vzero;

  const bf16* xrow = X + (size_t)(m0 + c15) * HDIM;
  for (int k0 = 0; k0 < HDIM; k0 += 32) {
    v8bf lo = *(const v8bf*)(xrow + k0 + half16);
    v8bf hi = *(const v8bf*)(xrow + k0 + half16 + 8);
    v16bf a = combine8(lo, hi);
#pragma unroll
    for (int t = 0; t < 4; ++t) {
      const bf16* wr = W + (size_t)(n0 + t * 16 + c15) * HDIM + k0 + half16;
      v16bf b = *(const v16bf*)wr;
      acc[t] = wmma_bf16(a, b, acc[t]);
    }
  }

#pragma unroll
  for (int t = 0; t < 4; ++t) {
    const int col = n0 + t * 16 + c15;
    const float bv = bias[col];
    const int h = col >> 6, d = col & 63;
#pragma unroll
    for (int r = 0; r < 8; ++r) {
      const int row = m0 + r + base8;
      const int bb = row >> 12, s = row & (SEQ - 1);
      const int bh = bb * NH + h;
      const float v = acc[t][r] + bv;
      const size_t idx = vmode ? ((size_t)bh * HD + d) * SEQ + s
                               : ((size_t)bh * SEQ + s) * HD + d;
      out[idx] = (bf16)v;
    }
  }
}

// ---------------------------------------------------------------------------
// Flash attention with async-DMA double-buffered K/V staging in LDS.
// grid (SEQ/64, BSZ*NH), block 128 (4 waves); wave handles one 16-query tile,
// all 4 waves share the 32-key K/V tiles staged in LDS.
// ---------------------------------------------------------------------------
__global__ __launch_bounds__(128) void attn_kernel(
    const bf16* __restrict__ Q, const bf16* __restrict__ K,
    const bf16* __restrict__ Vt, const float* __restrict__ mask,
    bf16* __restrict__ ctx) {
  // double-buffered tiles: K tile [key][dim] 32x64, V tile [dim][key] 64x32
  __shared__ __align__(32) bf16 kt[2][32 * HD];
  __shared__ __align__(32) bf16 vt[2][HD * 32];
  __shared__ __align__(32) bf16 plds[4][16 * 32];

  const int tid    = threadIdx.x;
  const int lane   = tid & 31;
  const int wv     = tid >> 5;
  const int c15    = lane & 15;
  const int half16 = (lane < 16) ? 0 : 16;
  const int base8  = (lane < 16) ? 0 : 8;
  const int bh = blockIdx.y;                    // 0..31
  const int q0 = (blockIdx.x * 4 + wv) * 16;    // seq-local query tile base

  // Q A-fragments for head_dim halves [0,32) and [32,64)
  const bf16* qrow = Q + ((size_t)bh * SEQ + q0 + c15) * HD;
  v16bf qa[2];
#pragma unroll
  for (int kk = 0; kk < 2; ++kk) {
    v8bf lo = *(const v8bf*)(qrow + kk * 32 + half16);
    v8bf hi = *(const v8bf*)(qrow + kk * 32 + half16 + 8);
    qa[kk] = combine8(lo, hi);
  }

  float mi[8], li[8];
#pragma unroll
  for (int r = 0; r < 8; ++r) { mi[r] = -INFINITY; li[r] = 0.0f; }

  const v8f vzero = {};
  v8f o[4];
#pragma unroll
  for (int t = 0; t < 4; ++t) o[t] = vzero;

  const float scale = 0.125f;  // 1/sqrt(64)

  // ---- async stage of one 32-key block into LDS buffer `buf` -------------
  const int vrow  = tid >> 1;          // 0..63  (dim row of V tile)
  const int vpart = (tid & 1) * 16;    // 0 or 16 (key half)
  auto issue_tile = [&](int j0, int buf) {
    // K tile: 32 keys x 64 dims, fully contiguous 4KB in global
    const bf16* kg = K + ((size_t)bh * SEQ + j0) * HD + tid * 16;
    async_copy_b128(kg,     &kt[buf][tid * 16]);
    async_copy_b128(kg + 8, &kt[buf][tid * 16 + 8]);
    // V tile: 64 dim-rows x 32 keys (Vt layout [bh][d][s])
    const bf16* vg = Vt + ((size_t)bh * HD + vrow) * SEQ + j0 + vpart;
    async_copy_b128(vg,     &vt[buf][vrow * 32 + vpart]);
    async_copy_b128(vg + 8, &vt[buf][vrow * 32 + vpart + 8]);
  };

  issue_tile(0, 0);

  const int NB = SEQ / 32;
  for (int jb = 0; jb < NB; ++jb) {
    const int j0  = jb * 32;
    const int buf = jb & 1;

    wait_async0();       // our async writes for `buf` (and older) landed
    __syncthreads();     // everyone's writes landed; prev reads of buf^1 done

    if (jb + 1 < NB) issue_tile(j0 + 32, buf ^ 1);  // prefetch next block

    // ---- scores: 16 queries x 32 keys (fragments from LDS) --------------
    v8f sc0 = vzero, sc1 = vzero;
#pragma unroll
    for (int kk = 0; kk < 2; ++kk) {
      v16bf b0 = *(const v16bf*)&kt[buf][(c15)      * HD + kk * 32 + half16];
      v16bf b1 = *(const v16bf*)&kt[buf][(16 + c15) * HD + kk * 32 + half16];
      sc0 = wmma_bf16(qa[kk], b0, sc0);
      sc1 = wmma_bf16(qa[kk], b1, sc1);
    }

    // ---- online softmax (rows r / r+8 split across lane halves) ---------
    float p0[8], p1[8], alpha[8];
#pragma unroll
    for (int r = 0; r < 8; ++r) {
      const int row = q0 + r + base8;
      float s0 = sc0[r] * scale + mask[(size_t)row * SEQ + j0 + c15];
      float s1 = sc1[r] * scale + mask[(size_t)row * SEQ + j0 + 16 + c15];
      float v = fmaxf(s0, s1);
      v = fmaxf(v, __shfl_xor(v, 1, 32));
      v = fmaxf(v, __shfl_xor(v, 2, 32));
      v = fmaxf(v, __shfl_xor(v, 4, 32));
      v = fmaxf(v, __shfl_xor(v, 8, 32));
      const float mnew = fmaxf(mi[r], v);
      const float a = __expf(mi[r] - mnew);
      mi[r] = mnew;
      const float e0 = __expf(s0 - mnew);
      const float e1 = __expf(s1 - mnew);
      float rs = e0 + e1;
      rs += __shfl_xor(rs, 1, 32);
      rs += __shfl_xor(rs, 2, 32);
      rs += __shfl_xor(rs, 4, 32);
      rs += __shfl_xor(rs, 8, 32);
      li[r] = li[r] * a + rs;
      alpha[r] = a;
      p0[r] = e0;
      p1[r] = e1;
    }

    // rescale running output
#pragma unroll
    for (int t = 0; t < 4; ++t)
#pragma unroll
      for (int r = 0; r < 8; ++r) o[t][r] *= alpha[r];

    // ---- transpose P (16x32) through per-wave LDS into an A-fragment ----
#pragma unroll
    for (int r = 0; r < 8; ++r) {
      plds[wv][(r + base8) * 32 + c15]      = (bf16)p0[r];
      plds[wv][(r + base8) * 32 + 16 + c15] = (bf16)p1[r];
    }
    __builtin_amdgcn_wave_barrier();  // keep ds stores before ds loads
    v8bf plo = *(const v8bf*)&plds[wv][c15 * 32 + half16];
    v8bf phi = *(const v8bf*)&plds[wv][c15 * 32 + half16 + 8];
    __builtin_amdgcn_wave_barrier();  // keep loads before next iter's stores
    v16bf pa = combine8(plo, phi);

    // ---- O += P * V  (V tile in LDS, contiguous along keys) -------------
#pragma unroll
    for (int t = 0; t < 4; ++t) {
      v16bf vb = *(const v16bf*)&vt[buf][(t * 16 + c15) * 32 + half16];
      o[t] = wmma_bf16(pa, vb, o[t]);
    }
  }

  // normalize and store context as [b][s][H] bf16
  const int bb = bh >> 4, h = bh & 15;
#pragma unroll
  for (int t = 0; t < 4; ++t) {
#pragma unroll
    for (int r = 0; r < 8; ++r) {
      const int row = q0 + r + base8;
      const int col = h * HD + t * 16 + c15;
      const float val = o[t][r] / li[r];
      ctx[((size_t)bb * SEQ + row) * HDIM + col] = (bf16)val;
    }
  }
}

// ---------------------------------------------------------------------------
// Output projection: out(f32) = ctx(bf16) * Wo^T + bo
// ---------------------------------------------------------------------------
__global__ __launch_bounds__(128) void out_proj_kernel(
    const bf16* __restrict__ X, const bf16* __restrict__ W,
    const float* __restrict__ bias, float* __restrict__ out) {
  const int lane   = threadIdx.x & 31;
  const int wv     = threadIdx.x >> 5;
  const int c15    = lane & 15;
  const int half16 = (lane < 16) ? 0 : 16;
  const int base8  = (lane < 16) ? 0 : 8;
  const int m0 = (blockIdx.x * 4 + wv) * 16;
  const int n0 = blockIdx.y * 64;

  const v8f vzero = {};
  v8f acc[4];
#pragma unroll
  for (int t = 0; t < 4; ++t) acc[t] = vzero;

  const bf16* xrow = X + (size_t)(m0 + c15) * HDIM;
  for (int k0 = 0; k0 < HDIM; k0 += 32) {
    v8bf lo = *(const v8bf*)(xrow + k0 + half16);
    v8bf hi = *(const v8bf*)(xrow + k0 + half16 + 8);
    v16bf a = combine8(lo, hi);
#pragma unroll
    for (int t = 0; t < 4; ++t) {
      const bf16* wr = W + (size_t)(n0 + t * 16 + c15) * HDIM + k0 + half16;
      v16bf b = *(const v16bf*)wr;
      acc[t] = wmma_bf16(a, b, acc[t]);
    }
  }

#pragma unroll
  for (int t = 0; t < 4; ++t) {
    const int col = n0 + t * 16 + c15;
    const float bv = bias[col];
#pragma unroll
    for (int r = 0; r < 8; ++r) {
      const int row = m0 + r + base8;
      out[(size_t)row * HDIM + col] = acc[t][r] + bv;
    }
  }
}

// ---------------------------------------------------------------------------
// launcher
// ---------------------------------------------------------------------------
extern "C" void kernel_launch(void* const* d_in, const int* in_sizes, int n_in,
                              void* d_out, int out_size, void* d_ws,
                              size_t ws_size, hipStream_t stream) {
  const float* x    = (const float*)d_in[0];
  const float* mask = (const float*)d_in[1];
  const float* wq   = (const float*)d_in[2];
  const float* bq   = (const float*)d_in[3];
  const float* wk   = (const float*)d_in[4];
  const float* bk   = (const float*)d_in[5];
  const float* wv   = (const float*)d_in[6];
  const float* bv   = (const float*)d_in[7];
  const float* wo   = (const float*)d_in[8];
  const float* bo   = (const float*)d_in[9];
  float* out = (float*)d_out;

  const size_t SZ_X = (size_t)MTOT * HDIM;  // 8,388,608 elems
  const size_t SZ_W = (size_t)HDIM * HDIM;  // 1,048,576 elems

  bf16* ws  = (bf16*)d_ws;
  bf16* xb  = ws;
  bf16* wqb = xb + SZ_X;
  bf16* wkb = wqb + SZ_W;
  bf16* wvb = wkb + SZ_W;
  bf16* wob = wvb + SZ_W;
  bf16* qb  = wob + SZ_W;
  bf16* kb  = qb + SZ_X;
  bf16* vtb = kb + SZ_X;
  bf16* ctx = vtb + SZ_X;

  // 1) downconvert activations + weights to bf16
  {
    int n4 = (int)(SZ_X / 4);
    cvt_f32_bf16_kernel<<<(n4 + 255) / 256, 256, 0, stream>>>(
        (const float4*)x, (v4bf*)xb, n4);
    int w4 = (int)(SZ_W / 4);
    cvt_f32_bf16_kernel<<<(w4 + 255) / 256, 256, 0, stream>>>(
        (const float4*)wq, (v4bf*)wqb, w4);
    cvt_f32_bf16_kernel<<<(w4 + 255) / 256, 256, 0, stream>>>(
        (const float4*)wk, (v4bf*)wkb, w4);
    cvt_f32_bf16_kernel<<<(w4 + 255) / 256, 256, 0, stream>>>(
        (const float4*)wv, (v4bf*)wvb, w4);
    cvt_f32_bf16_kernel<<<(w4 + 255) / 256, 256, 0, stream>>>(
        (const float4*)wo, (v4bf*)wob, w4);
  }

  // 2) QKV projections (WMMA GEMMs)
  {
    dim3 grid(MTOT / 64, HDIM / 64);
    qkv_gemm_kernel<<<grid, 128, 0, stream>>>(xb, wqb, bq, qb, 0);
    qkv_gemm_kernel<<<grid, 128, 0, stream>>>(xb, wkb, bk, kb, 0);
    qkv_gemm_kernel<<<grid, 128, 0, stream>>>(xb, wvb, bv, vtb, 1);
  }

  // 3) flash attention (async double-buffered K/V staging)
  {
    dim3 grid(SEQ / 64, BSZ * NH);
    attn_kernel<<<grid, 128, 0, stream>>>(qb, kb, vtb, mask, ctx);
  }

  // 4) output projection -> f32 d_out
  {
    dim3 grid(MTOT / 64, HDIM / 64);
    out_proj_kernel<<<grid, 128, 0, stream>>>(ctx, wob, bo, out);
  }

  (void)in_sizes; (void)n_in; (void)out_size; (void)ws_size;
}
